// Tacotron2_8667244003338
// MI455X (gfx1250) — matrix operand order repaired and checked by reading, compiled, not run on hardware
//
#include <hip/hip_runtime.h>
#include <hip/hip_bf16.h>

// ---------------------------------------------------------------------------
// Tacotron2 decoder for MI455X (gfx1250, wave32, WMMA).
// All weights converted to bf16 once per launch (L2-resident, 192MB, across
// the 128 sequential decoder steps). Every GEMM runs v_wmma_f32_16x16x32_bf16
// with M=16 (= batch) per fragment; activations staged/converted in LDS;
// weights read as contiguous 32B per lane from L2 via GLOBAL loads. Each wave
// owns TWO independent accumulators (32 output columns); K unrolled by 2
// (4 WMMAs/iter) with pointer-incremented addressing.
// ---------------------------------------------------------------------------

#define BB    16
#define TT    128
#define LL    128
#define IDIM  512
#define ODIM  80
#define DUNITS 1024
#define PRE   256
#define ADIM  128
#define ACONV 32
#define AKW   31
#define PCH   512
#define PKW   5
#define EPSV  1e-5f

typedef __attribute__((ext_vector_type(16))) __bf16 v16bf;
typedef __attribute__((ext_vector_type(8)))  __bf16 v8bf;
typedef __attribute__((ext_vector_type(8)))  float  v8f;

union AFrag { v16bf v; v8bf h[2]; };

#define WMMA_BF16(a, b, c) \
  __builtin_amdgcn_wmma_f32_16x16x32_bf16(false, (a), false, (b), (short)0, (c), false, false)

// ---------------------------------------------------------------------------
// One GEMM K-segment: accumulate X_s @ W_s^T into acc0/acc1.
// Passed kernarg pointers directly (keeps global addrspace inference ->
// global_load_b128, not flat_load_b128).  kp is block-uniform.
// ---------------------------------------------------------------------------
static __device__ __forceinline__ void gemm_seg(
    const float* __restrict__ xp, int ldx, const __bf16* __restrict__ wb, int ldw,
    int kr, int kp, int m0, int tid, int l16, int lhalf, int nb0, int nb1,
    __bf16* sx, v8f& acc0, v8f& acc1)
{
  for (int kc = 0; kc < kp; kc += 1024) {
    int kchunk = kp - kc; if (kchunk > 1024) kchunk = 1024;
    // stage + fp32->bf16 convert the 16-row activation panel for this chunk
    if (kchunk == 1024) {
      for (int i = tid; i < 16 * 1024; i += 128) {
        int row = i >> 10, cl = i & 1023;
        int lc = kc + cl;
        float v = (lc < kr) ? xp[(size_t)(m0 + row) * ldx + lc] : 0.f;
        sx[row * 1024 + cl] = (__bf16)v;
      }
    } else {
      for (int i = tid; i < 16 * kchunk; i += 128) {
        int row = i / kchunk, cl = i - row * kchunk;
        int lc = kc + cl;
        float v = (lc < kr) ? xp[(size_t)(m0 + row) * ldx + lc] : 0.f;
        sx[row * 1024 + cl] = (__bf16)v;
      }
    }
    __syncthreads();

    // per-lane weight row pointers (each k-tile = contiguous 32B per lane)
    const __bf16* wr0 = wb + (size_t)(nb0 + l16) * ldw + kc + 16 * lhalf;
    const __bf16* wr1 = wb + (size_t)(nb1 + l16) * ldw + kc + 16 * lhalf;
    if (kc + 1024 < kp) {                 // prefetch next chunk (L2 -> near)
      __builtin_prefetch(wr0 + 1024, 0, 1);
      __builtin_prefetch(wr1 + 1024, 0, 1);
    }
    const __bf16* ar = &sx[l16 * 1024 + 16 * lhalf];

    int k0 = 0;
    for (; k0 + 64 <= kchunk; k0 += 64) {   // 2 K-tiles, 4 independent WMMAs
      AFrag a0, a1;
      a0.h[0] = *(const v8bf*)(ar);
      a0.h[1] = *(const v8bf*)(ar + 8);
      a1.h[0] = *(const v8bf*)(ar + 32);
      a1.h[1] = *(const v8bf*)(ar + 40);
      v16bf b00 = *(const v16bf*)(wr0);
      v16bf b10 = *(const v16bf*)(wr1);
      v16bf b01 = *(const v16bf*)(wr0 + 32);
      v16bf b11 = *(const v16bf*)(wr1 + 32);
      acc0 = WMMA_BF16(a0.v, b00, acc0);
      acc1 = WMMA_BF16(a0.v, b10, acc1);
      acc0 = WMMA_BF16(a1.v, b01, acc0);
      acc1 = WMMA_BF16(a1.v, b11, acc1);
      ar += 64; wr0 += 64; wr1 += 64;
    }
    if (k0 < kchunk) {                      // one remaining 32-K tile
      AFrag a0;
      a0.h[0] = *(const v8bf*)(ar);
      a0.h[1] = *(const v8bf*)(ar + 8);
      v16bf b00 = *(const v16bf*)(wr0);
      v16bf b10 = *(const v16bf*)(wr1);
      acc0 = WMMA_BF16(a0.v, b00, acc0);
      acc1 = WMMA_BF16(a0.v, b10, acc1);
    }
    __syncthreads();
  }
}

// ---------------------------------------------------------------------------
// Generic WMMA GEMM: Y[m0+0..15][0..N) = act( sum_seg X_s @ W_s^T + biases )
//   X_s : fp32 [*, ldx_s] (rows m0..m0+15), converted to bf16 in LDS
//   W_s : bf16 [N, ldw_s] row-major (i.e. transposed weight), L2-resident
// 4 waves/block, 2 16x16 tiles (32 cols) per wave.
// ---------------------------------------------------------------------------
__global__ __launch_bounds__(128)
void gemm_bf16_wmma(
    const float* __restrict__ x0, int ldx0, const __bf16* __restrict__ w0, int ldw0, int kr0, int kp0,
    const float* __restrict__ x1, int ldx1, const __bf16* __restrict__ w1, int ldw1, int kr1, int kp1,
    const float* __restrict__ x2, int ldx2, const __bf16* __restrict__ w2, int ldw2, int kr2, int kp2,
    const float* __restrict__ bias0, const float* __restrict__ bias1,
    float* __restrict__ y, int ldy, int N, int act)
{
  __shared__ __bf16 sx[16 * 1024];
  const int tid   = threadIdx.x;
  const int wv    = tid >> 5;
  const int lane  = tid & 31;
  const int l16   = lane & 15;
  const int lhalf = lane >> 4;
  const int m0    = blockIdx.y * 16;
  const int n0    = (blockIdx.x * 4 + wv) * 32;       // 2 tiles per wave
  const bool act0 = (n0 < N);
  const bool act1 = (n0 + 16 < N);
  const int  nb0  = act0 ? n0 : 0;                    // clamped: safe loads
  const int  nb1  = act1 ? (n0 + 16) : 0;

  v8f acc0 = {0.f, 0.f, 0.f, 0.f, 0.f, 0.f, 0.f, 0.f};
  v8f acc1 = {0.f, 0.f, 0.f, 0.f, 0.f, 0.f, 0.f, 0.f};

  if (kp0) gemm_seg(x0, ldx0, w0, ldw0, kr0, kp0, m0, tid, l16, lhalf, nb0, nb1, sx, acc0, acc1);
  if (kp1) gemm_seg(x1, ldx1, w1, ldw1, kr1, kp1, m0, tid, l16, lhalf, nb0, nb1, sx, acc0, acc1);
  if (kp2) gemm_seg(x2, ldx2, w2, ldw2, kr2, kp2, m0, tid, l16, lhalf, nb0, nb1, sx, acc0, acc1);

  // C/D layout: VGPR vi, lanes 0-15 -> M=vi, lanes 16-31 -> M=vi+8
  for (int vi = 0; vi < 8; ++vi) {
    int row = vi + 8 * lhalf;
    if (act0) {
      int col = n0 + l16;
      float r = acc0[vi];
      if (bias0) r += bias0[col];
      if (bias1) r += bias1[col];
      if (act == 1) r = fmaxf(r, 0.f);
      y[(size_t)(m0 + row) * ldy + col] = r;
    }
    if (act1) {
      int col = n0 + 16 + l16;
      float r = acc1[vi];
      if (bias0) r += bias0[col];
      if (bias1) r += bias1[col];
      if (act == 1) r = fmaxf(r, 0.f);
      y[(size_t)(m0 + row) * ldy + col] = r;
    }
  }
}

// ---------------------------------------------------------------------------
// one-time weight conversion kernels
// ---------------------------------------------------------------------------
__global__ void convert_rm(const float* __restrict__ w, __bf16* __restrict__ o,
                           int N, int K, int ldo) {
  // w [N,K] row-major -> o [N,ldo] bf16, zero pad K..ldo
  for (size_t i = blockIdx.x * blockDim.x + threadIdx.x; i < (size_t)N * ldo;
       i += (size_t)gridDim.x * blockDim.x) {
    int n = (int)(i / ldo), c = (int)(i % ldo);
    o[i] = (c < K) ? (__bf16)w[(size_t)n * K + c] : (__bf16)0.f;
  }
}

__global__ void convert_tr(const float* __restrict__ w, __bf16* __restrict__ o,
                           int K, int N, int ldo) {
  // w [K,N] row-major -> o [N,ldo] bf16 (transpose), zero pad K..ldo
  for (size_t i = blockIdx.x * blockDim.x + threadIdx.x; i < (size_t)N * ldo;
       i += (size_t)gridDim.x * blockDim.x) {
    int n = (int)(i / ldo), c = (int)(i % ldo);
    o[i] = (c < K) ? (__bf16)w[(size_t)c * N + n] : (__bf16)0.f;
  }
}

__global__ void zero_f32(float* p, int n) {
  for (int i = blockIdx.x * blockDim.x + threadIdx.x; i < n; i += gridDim.x * blockDim.x)
    p[i] = 0.f;
}

// ---------------------------------------------------------------------------
// attention: per (b,t) energy e = gvec . tanh(enc_proj + att_conv + dec_proj)
// ---------------------------------------------------------------------------
__global__ __launch_bounds__(128)
void attn_energy(const float* __restrict__ encp,   // [B*T, ADIM]
                 const float* __restrict__ dp,     // [B, ADIM]
                 const float* __restrict__ cum_w,  // [B, T]
                 const int*   __restrict__ hlens,
                 const float* __restrict__ convk,  // [ACONV, AK]
                 const float* __restrict__ mlpw,   // [ACONV, ADIM]
                 const float* __restrict__ gvec, const float* __restrict__ gvecb,
                 float* __restrict__ e, int step)
{
  int b = blockIdx.x;
  int t = threadIdx.x;
  int hl = hlens[b];
  __shared__ float ap[TT];
  float apv;
  if (step == 0) apv = (t < hl) ? (1.0f / (float)hl) : 0.f;
  else           apv = cum_w[b * TT + t];
  ap[t] = apv;
  __syncthreads();

  // 1D conv, 32 channels, kernel 31, pad 15
  float conv[ACONV];
#pragma unroll
  for (int c = 0; c < ACONV; ++c) conv[c] = 0.f;
  for (int k = 0; k < AKW; ++k) {
    int tt = t + k - 15;
    if (tt >= 0 && tt < TT) {
      float v = ap[tt];
#pragma unroll
      for (int c = 0; c < ACONV; ++c) conv[c] += convk[c * AKW + k] * v;
    }
  }

  const float* ep = encp + (size_t)(b * TT + t) * ADIM;
  const float* dpb = dp + b * ADIM;
  float s = 0.f;
  for (int a = 0; a < ADIM; ++a) {
    float v = ep[a] + dpb[a];
#pragma unroll
    for (int c = 0; c < ACONV; ++c) v += conv[c] * mlpw[c * ADIM + a];
    s += gvec[a] * tanhf(v);
  }
  s += gvecb[0];
  e[b * TT + t] = (t < hl) ? s : -1e30f;
}

// softmax (SCALING=2) + context + att_ws output + cum_w update
__global__ __launch_bounds__(128)
void softmax_ctx(const float* __restrict__ e, const float* __restrict__ hs,
                 float* __restrict__ cum_w, float* __restrict__ att_c,
                 float* __restrict__ attws, int step)
{
  int b = blockIdx.x;
  int t = threadIdx.x;
  __shared__ float sm[TT];
  __shared__ float wb[TT];
  float ev = 2.0f * e[b * TT + t];
  sm[t] = ev; __syncthreads();
  for (int s = 64; s > 0; s >>= 1) { if (t < s) sm[t] = fmaxf(sm[t], sm[t + s]); __syncthreads(); }
  float mx = sm[0]; __syncthreads();
  float ex = expf(ev - mx);
  sm[t] = ex; __syncthreads();
  for (int s = 64; s > 0; s >>= 1) { if (t < s) sm[t] += sm[t + s]; __syncthreads(); }
  float w = ex / sm[0];
  wb[t] = w; __syncthreads();
  attws[((size_t)b * LL + step) * TT + t] = w;
  cum_w[b * TT + t] += w;
  for (int d = t; d < IDIM; d += TT) {
    float s2 = 0.f;
    for (int tt = 0; tt < TT; ++tt) s2 += wb[tt] * hs[(size_t)(b * TT + tt) * IDIM + d];
    att_c[b * IDIM + d] = s2;
  }
}

// LSTM gate nonlinearity (torch gate order i,f,g,o)
__global__ __launch_bounds__(256)
void lstm_point(const float* __restrict__ gates, float* __restrict__ h, float* __restrict__ c)
{
  int i = blockIdx.x * blockDim.x + threadIdx.x;   // 16*1024
  int b = i >> 10, u = i & 1023;
  const float* g = gates + (size_t)b * 4 * DUNITS;
  float gi = g[u], gf = g[DUNITS + u], gg = g[2 * DUNITS + u], go = g[3 * DUNITS + u];
  float si = 1.f / (1.f + expf(-gi));
  float sf = 1.f / (1.f + expf(-gf));
  float so = 1.f / (1.f + expf(-go));
  float c2 = sf * c[i] + si * tanhf(gg);
  h[i] = so * tanhf(c2);
  c[i] = c2;
}

// stop-token logit: dot(concat(z1, att_c), prob_w) + b
__global__ __launch_bounds__(256)
void prob_kernel(const float* __restrict__ z1, const float* __restrict__ att_c,
                 const float* __restrict__ pw, const float* __restrict__ pb,
                 float* __restrict__ logits, int step)
{
  int b = blockIdx.x, t = threadIdx.x;
  float s = 0.f;
  for (int k = t; k < DUNITS; k += 256) s += z1[b * DUNITS + k] * pw[k];
  for (int k = t; k < IDIM; k += 256)   s += att_c[b * IDIM + k] * pw[DUNITS + k];
  __shared__ float red[256];
  red[t] = s; __syncthreads();
  for (int st = 128; st > 0; st >>= 1) { if (t < st) red[t] += red[t + st]; __syncthreads(); }
  if (t == 0) logits[b * LL + step] = red[0] + pb[0];
}

// ---------------------------------------------------------------------------
// postnet helpers
// ---------------------------------------------------------------------------
__global__ void transpose_blc(const float* __restrict__ before, float* __restrict__ x)
{ // before [B,L,80] -> x [B,80,L]
  for (int i = blockIdx.x * blockDim.x + threadIdx.x; i < BB * LL * ODIM;
       i += gridDim.x * blockDim.x) {
    int b = i / (LL * ODIM), r = i % (LL * ODIM), l = r / ODIM, c = r % ODIM;
    x[((size_t)b * ODIM + c) * LL + l] = before[i];
  }
}

__global__ void im2col_k(const float* __restrict__ x, float* __restrict__ col, int C, int ldc)
{ // x [B,C,L] -> col [B*L, C*5] (pad 2), fp32
  size_t total = (size_t)BB * LL * C * PKW;
  for (size_t i = blockIdx.x * (size_t)blockDim.x + threadIdx.x; i < total;
       i += (size_t)gridDim.x * blockDim.x) {
    int j  = (int)(i % (C * PKW));
    int m  = (int)(i / (C * PKW));
    int ci = j / PKW, k = j % PKW;
    int b = m >> 7, l = m & 127;
    int ll = l + k - 2;
    float v = (ll >= 0 && ll < LL) ? x[((size_t)b * C + ci) * LL + ll] : 0.f;
    col[(size_t)m * ldc + j] = v;
  }
}

__global__ __launch_bounds__(256)
void bn_stats(const float* __restrict__ y, int ldy, float* __restrict__ mean, float* __restrict__ var)
{
  int co = blockIdx.x, t = threadIdx.x;
  float s = 0.f, s2 = 0.f;
  for (int m = t; m < BB * LL; m += 256) { float v = y[(size_t)m * ldy + co]; s += v; s2 += v * v; }
  __shared__ float r1[256], r2[256];
  r1[t] = s; r2[t] = s2; __syncthreads();
  for (int st = 128; st > 0; st >>= 1) { if (t < st) { r1[t] += r1[t + st]; r2[t] += r2[t + st]; } __syncthreads(); }
  if (t == 0) { float mu = r1[0] / (BB * LL); mean[co] = mu; var[co] = r2[0] / (BB * LL) - mu * mu; }
}

__global__ void bn_apply(const float* __restrict__ y, int ldy,
                         const float* __restrict__ mean, const float* __restrict__ var,
                         const float* __restrict__ g, const float* __restrict__ beta,
                         float* __restrict__ xout, int C)
{ // xout [B,C,L], tanh activation
  for (int i = blockIdx.x * blockDim.x + threadIdx.x; i < BB * C * LL;
       i += gridDim.x * blockDim.x) {
    int l = i % LL, bc = i / LL, co = bc % C, b = bc / C;
    int m = b * LL + l;
    float v = (y[(size_t)m * ldy + co] - mean[co]) * rsqrtf(var[co] + EPSV) * g[co] + beta[co];
    xout[i] = tanhf(v);
  }
}

__global__ void bn_apply_last(const float* __restrict__ y, int ldy,
                              const float* __restrict__ mean, const float* __restrict__ var,
                              const float* __restrict__ g, const float* __restrict__ beta,
                              const float* __restrict__ before, float* __restrict__ after)
{ // after [B,L,80] = before + bn(y)
  for (int i = blockIdx.x * blockDim.x + threadIdx.x; i < BB * LL * ODIM;
       i += gridDim.x * blockDim.x) {
    int c = i % ODIM, bl = i / ODIM;
    float v = (y[(size_t)bl * ldy + c] - mean[c]) * rsqrtf(var[c] + EPSV) * g[c] + beta[c];
    after[i] = before[i] + v;
  }
}

// ---------------------------------------------------------------------------
// host side
// ---------------------------------------------------------------------------
static inline void gemm_launch(hipStream_t s,
    const float* x0, int ldx0, const __bf16* w0, int ldw0, int kr0, int kp0,
    const float* x1, int ldx1, const __bf16* w1, int ldw1, int kr1, int kp1,
    const float* x2, int ldx2, const __bf16* w2, int ldw2, int kr2, int kp2,
    const float* b0, const float* b1, float* y, int ldy, int N, int mtiles, int act)
{
  dim3 grid((N + 127) / 128, mtiles);   // 4 waves x 32 cols per block
  gemm_bf16_wmma<<<grid, dim3(128), 0, s>>>(x0, ldx0, w0, ldw0, kr0, kp0,
                                            x1, ldx1, w1, ldw1, kr1, kp1,
                                            x2, ldx2, w2, ldw2, kr2, kp2,
                                            b0, b1, y, ldy, N, act);
}

extern "C" void kernel_launch(void* const* d_in, const int* in_sizes, int n_in,
                              void* d_out, int out_size, void* d_ws, size_t ws_size,
                              hipStream_t stream) {
  // ---- inputs (setup_inputs dict order, params dict flattened in order) ----
  const float* hs        = (const float*)d_in[0];   // [16,128,512]
  const int*   hlens     = (const int*)  d_in[1];   // [16]
  const float* ys        = (const float*)d_in[2];   // [16,128,80]
  const float* prenet_w0 = (const float*)d_in[3];   // [80,256]
  const float* prenet_w1 = (const float*)d_in[4];   // [256,256]
  const float* l0_wih    = (const float*)d_in[5];   // [4096,768]
  const float* l0_whh    = (const float*)d_in[6];   // [4096,1024]
  const float* l0_bih    = (const float*)d_in[7];
  const float* l0_bhh    = (const float*)d_in[8];
  const float* l1_wih    = (const float*)d_in[9];   // [4096,1024]
  const float* l1_whh    = (const float*)d_in[10];  // [4096,1024]
  const float* l1_bih    = (const float*)d_in[11];
  const float* l1_bhh    = (const float*)d_in[12];
  const float* enc_w     = (const float*)d_in[13];  // [512,128]
  const float* enc_b     = (const float*)d_in[14];  // [128]
  const float* dec_w     = (const float*)d_in[15];  // [1024,128]
  const float* conv_k    = (const float*)d_in[16];  // [32,1,31]
  const float* mlp_w     = (const float*)d_in[17];  // [32,128]
  const float* gvec_w    = (const float*)d_in[18];  // [128]
  const float* gvec_b    = (const float*)d_in[19];  // scalar
  const float* feat_w    = (const float*)d_in[20];  // [1536,80]
  const float* prob_w    = (const float*)d_in[21];  // [1536,1]
  const float* prob_b    = (const float*)d_in[22];  // [1]
  const float* post_k[5], *post_g[5], *post_bb[5];
  for (int i = 0; i < 5; ++i) {
    post_k[i]  = (const float*)d_in[23 + 3 * i];
    post_g[i]  = (const float*)d_in[24 + 3 * i];
    post_bb[i] = (const float*)d_in[25 + 3 * i];
  }

  // ---- outputs: after, before, logits, att_ws (concatenated fp32) ----
  float* d_after  = (float*)d_out;
  float* d_before = d_after + BB * LL * ODIM;
  float* d_logits = d_before + BB * LL * ODIM;
  float* d_attws  = d_logits + BB * LL;

  // ---- workspace carve ----
  size_t off = 0;
  char* base = (char*)d_ws;
  auto carve = [&](size_t bytes) -> void* {
    void* p = base + off; off += (bytes + 255) & ~(size_t)255; return p;
  };
  __bf16* bW_ih0 = (__bf16*)carve((size_t)4096 * 768 * 2);
  __bf16* bW_hh0 = (__bf16*)carve((size_t)4096 * 1024 * 2);
  __bf16* bW_ih1 = (__bf16*)carve((size_t)4096 * 1024 * 2);
  __bf16* bW_hh1 = (__bf16*)carve((size_t)4096 * 1024 * 2);
  __bf16* bPW0T  = (__bf16*)carve((size_t)256 * 96 * 2);
  __bf16* bPW1T  = (__bf16*)carve((size_t)256 * 256 * 2);
  __bf16* bENCT  = (__bf16*)carve((size_t)128 * 512 * 2);
  __bf16* bDECT  = (__bf16*)carve((size_t)128 * 1024 * 2);
  __bf16* bFOT   = (__bf16*)carve((size_t)80 * 1536 * 2);
  __bf16* bPK[5];
  for (int i = 0; i < 5; ++i) {
    int Cout = (i == 4) ? ODIM : PCH;
    bPK[i] = (__bf16*)carve((size_t)Cout * 2560 * 2);
  }
  float* encp   = (float*)carve((size_t)BB * TT * ADIM * 4);
  // zero-init state region (contiguous): z0,c0,z1,c1,cum_w,zeros80
  float* z0     = (float*)carve((4 * BB * DUNITS + BB * TT + BB * ODIM) * 4);
  float* c0     = z0 + BB * DUNITS;
  float* z1     = c0 + BB * DUNITS;
  float* c1     = z1 + BB * DUNITS;
  float* cum_w  = c1 + BB * DUNITS;
  float* zeros80 = cum_w + BB * TT;
  int    nzero  = 4 * BB * DUNITS + BB * TT + BB * ODIM;
  float* att_c  = (float*)carve(BB * IDIM * 4);
  float* pre_t  = (float*)carve(BB * PRE * 4);
  float* pre_h  = (float*)carve(BB * PRE * 4);
  float* dp     = (float*)carve(BB * ADIM * 4);
  float* ebuf   = (float*)carve(BB * TT * 4);
  float* gates  = (float*)carve(BB * 4 * DUNITS * 4);
  float* xA     = (float*)carve((size_t)BB * PCH * LL * 4);
  float* xB     = (float*)carve((size_t)BB * PCH * LL * 4);
  float* col    = (float*)carve((size_t)BB * LL * 2560 * 4);
  float* ybuf   = (float*)carve((size_t)BB * LL * PCH * 4);
  float* meanb  = (float*)carve(PCH * 4);
  float* varb   = (float*)carve(PCH * 4);

  // ---- one-time conversions + state init ----
  zero_f32<<<256, 256, 0, stream>>>(z0, nzero);
  convert_rm<<<2048, 256, 0, stream>>>(l0_wih, bW_ih0, 4096, 768, 768);
  convert_rm<<<2048, 256, 0, stream>>>(l0_whh, bW_hh0, 4096, 1024, 1024);
  convert_rm<<<2048, 256, 0, stream>>>(l1_wih, bW_ih1, 4096, 1024, 1024);
  convert_rm<<<2048, 256, 0, stream>>>(l1_whh, bW_hh1, 4096, 1024, 1024);
  convert_tr<<<128, 256, 0, stream>>>(prenet_w0, bPW0T, 80, 256, 96);
  convert_tr<<<256, 256, 0, stream>>>(prenet_w1, bPW1T, 256, 256, 256);
  convert_tr<<<256, 256, 0, stream>>>(enc_w, bENCT, 512, 128, 512);
  convert_tr<<<512, 256, 0, stream>>>(dec_w, bDECT, 1024, 128, 1024);
  convert_tr<<<512, 256, 0, stream>>>(feat_w, bFOT, 1536, 80, 1536);
  for (int i = 0; i < 5; ++i) {
    int Cin = (i == 0) ? ODIM : PCH;
    int Cout = (i == 4) ? ODIM : PCH;
    convert_rm<<<2048, 256, 0, stream>>>(post_k[i], bPK[i], Cout, Cin * PKW, 2560);
  }

  // enc_proj = hs @ att_enc_w + b   (M=2048 tiles of 16)
  gemm_launch(stream, hs, IDIM, bENCT, 512, 512, 512,
              nullptr, 0, nullptr, 0, 0, 0, nullptr, 0, nullptr, 0, 0, 0,
              enc_b, nullptr, encp, ADIM, ADIM, TT * BB / 16, 0);

  // ---- sequential decoder loop ----
  for (int t = 0; t < LL; ++t) {
    // dec_proj = z0 @ att_dec_w   (uses z0 from previous step)
    gemm_launch(stream, z0, DUNITS, bDECT, 1024, 1024, 1024,
                nullptr, 0, nullptr, 0, 0, 0, nullptr, 0, nullptr, 0, 0, 0,
                nullptr, nullptr, dp, ADIM, ADIM, 1, 0);
    attn_energy<<<BB, TT, 0, stream>>>(encp, dp, cum_w, hlens, conv_k, mlp_w,
                                       gvec_w, gvec_b, ebuf, t);
    softmax_ctx<<<BB, TT, 0, stream>>>(ebuf, hs, cum_w, att_c, d_attws, t);

    // prenet(prev_out): prev_out = ys[:, t-1] (zeros at t=0)
    const float* pvx = (t == 0) ? zeros80 : (ys + (size_t)(t - 1) * ODIM);
    int pvld = (t == 0) ? ODIM : (LL * ODIM);
    gemm_launch(stream, pvx, pvld, bPW0T, 96, 80, 96,
                nullptr, 0, nullptr, 0, 0, 0, nullptr, 0, nullptr, 0, 0, 0,
                nullptr, nullptr, pre_t, PRE, PRE, 1, 1);
    gemm_launch(stream, pre_t, PRE, bPW1T, 256, 256, 256,
                nullptr, 0, nullptr, 0, 0, 0, nullptr, 0, nullptr, 0, 0, 0,
                nullptr, nullptr, pre_h, PRE, PRE, 1, 1);

    // LSTM0 gates: [att_c | pre_h] @ wih0^T + z0 @ whh0^T + biases
    gemm_launch(stream, att_c, IDIM, bW_ih0, 768, 512, 512,
                pre_h, PRE, bW_ih0 + 512, 768, 256, 256,
                z0, DUNITS, bW_hh0, 1024, 1024, 1024,
                l0_bih, l0_bhh, gates, 4 * DUNITS, 4 * DUNITS, 1, 0);
    lstm_point<<<(BB * DUNITS) / 256, 256, 0, stream>>>(gates, z0, c0);

    // LSTM1 gates: z0 @ wih1^T + z1 @ whh1^T + biases
    gemm_launch(stream, z0, DUNITS, bW_ih1, 1024, 1024, 1024,
                z1, DUNITS, bW_hh1, 1024, 1024, 1024,
                nullptr, 0, nullptr, 0, 0, 0,
                l1_bih, l1_bhh, gates, 4 * DUNITS, 4 * DUNITS, 1, 0);
    lstm_point<<<(BB * DUNITS) / 256, 256, 0, stream>>>(gates, z1, c1);

    // feat_out: [z1 | att_c] @ feat_out_w  -> before[:, t, :]
    gemm_launch(stream, z1, DUNITS, bFOT, 1536, 1024, 1024,
                att_c, IDIM, bFOT + 1024, 1536, 512, 512,
                nullptr, 0, nullptr, 0, 0, 0,
                nullptr, nullptr, d_before + (size_t)t * ODIM, LL * ODIM, ODIM, 1, 0);
    prob_kernel<<<BB, 256, 0, stream>>>(z1, att_c, prob_w, prob_b, d_logits, t);
  }

  // ---- postnet: 5x (conv1d as im2col GEMM + batch-stat BN + tanh) ----
  transpose_blc<<<640, 256, 0, stream>>>(d_before, xA);
  const float* xin = xA;
  float* xout = xB;
  for (int i = 0; i < 5; ++i) {
    int Cin  = (i == 0) ? ODIM : PCH;
    int Cout = (i == 4) ? ODIM : PCH;
    int Kreal = Cin * PKW;
    int Kpad  = (Kreal + 31) & ~31;
    im2col_k<<<2048, 256, 0, stream>>>(xin, col, Cin, 2560);
    gemm_launch(stream, col, 2560, bPK[i], 2560, Kreal, Kpad,
                nullptr, 0, nullptr, 0, 0, 0, nullptr, 0, nullptr, 0, 0, 0,
                nullptr, nullptr, ybuf, Cout, Cout, (BB * LL) / 16, 0);
    bn_stats<<<Cout, 256, 0, stream>>>(ybuf, Cout, meanb, varb);
    if (i < 4) {
      bn_apply<<<2048, 256, 0, stream>>>(ybuf, Cout, meanb, varb, post_g[i], post_bb[i], xout, Cout);
      const float* tmp = xin; xin = xout; xout = (float*)tmp;
    } else {
      bn_apply_last<<<640, 256, 0, stream>>>(ybuf, Cout, meanb, varb, post_g[i], post_bb[i],
                                             d_before, d_after);
    }
  }
}
